// MDN_RNN_46763603919160
// MI455X (gfx1250) — compile-verified, matrix operand order checked by reading, not run on hardware
//
#include <hip/hip_runtime.h>

// ---------------------------------------------------------------------------
// MDN-RNN for MI455X (gfx1250): bf16 WMMA everywhere, fp32 accumulate.
//  Phase 0: prep    — convert x -> bf16 time-major padded, pad MDN weights.
//  Phase 1: scan    — cooperative persistent LSTM (32 WGs, 1 grid barrier/step),
//                     c in registers, W_hh/W_ih slice in LDS, h ping-pong in L2.
//  Phase 2: 3 GEMMs — H_all -> hid1 -> hid2 -> o (bf16 WMMA, M = 261888 rows).
//  Phase 3: final   — softmax(coeff), exp(var), scatter to [B, T-1, ...].
// ---------------------------------------------------------------------------

#define B_SZ   256
#define T_SEQ  1024
#define TM1    1023
#define HID    256
#define IN_DIM 35
#define INPAD  64
#define G4     1024          // 4*HID gate rows
#define MROWS  (TM1 * B_SZ)  // 261888
#define HPAD   128           // MDN hidden (100) padded
#define OPAD   176           // MDN out (170) padded
#define NBLK_SCAN 32

typedef __attribute__((ext_vector_type(8)))  __bf16 v8bf;
typedef __attribute__((ext_vector_type(16))) __bf16 v16bf;
typedef __attribute__((ext_vector_type(8)))  float  v8f;

__device__ __forceinline__ unsigned short f2bf(float f) {
  unsigned u = __float_as_uint(f);
  u += 0x7FFFu + ((u >> 16) & 1u);          // round-to-nearest-even
  return (unsigned short)(u >> 16);
}

// Load one 16x32 bf16 A/B fragment chunk from a row-major [rows x ld] bf16
// matrix. Caller passes (base + row*ld + k0) where row = tile*16 + (lane&15)
// and k0 = kblock*32 + ((lane>>4)<<3). Two b128 loads: k..k+7 and k+16..k+23.
__device__ __forceinline__ v16bf ldfrag(const unsigned short* p) {
  const v8bf* q = reinterpret_cast<const v8bf*>(p);
  union { v16bf v; v8bf h[2]; } u;
  u.h[0] = q[0];
  u.h[1] = q[2];
  return u.v;
}

__device__ __forceinline__ v8f wmma_bf16(v16bf a, v16bf b, v8f c) {
  return __builtin_amdgcn_wmma_f32_16x16x32_bf16(false, a, false, b,
                                                 (short)0, c, false, false);
}

__device__ __forceinline__ float sigm(float x) {
  return 1.0f / (1.0f + __expf(-x));
}

// ---------------------------------------------------------------------------
// Phase 0: conversions / padding / zero-init
// ---------------------------------------------------------------------------
__global__ void __launch_bounds__(256) prep_kernel(
    const float* __restrict__ x,     // [B, T, 35]
    const float* __restrict__ W1, const float* __restrict__ b1,
    const float* __restrict__ W2, const float* __restrict__ b2,
    const float* __restrict__ W3, const float* __restrict__ b3,
    unsigned short* __restrict__ Xbf,   // [TM1][B][INPAD] bf16
    unsigned short* __restrict__ hbuf,  // 2 x [B][HID] bf16 (zeroed)
    unsigned short* __restrict__ W1p,   // [HPAD][256]
    unsigned short* __restrict__ W2p,   // [HPAD][HPAD]
    unsigned short* __restrict__ W3p,   // [OPAD][HPAD]
    float* __restrict__ b1p, float* __restrict__ b2p, float* __restrict__ b3p)
{
  const long long stride = (long long)gridDim.x * blockDim.x;
  const long long tid0   = (long long)blockIdx.x * blockDim.x + threadIdx.x;

  // x -> bf16, time-major, K padded 35 -> 64
  for (long long i = tid0; i < (long long)TM1 * B_SZ * INPAD; i += stride) {
    long long t = i / (B_SZ * INPAD);
    int b = (int)((i / INPAD) % B_SZ);
    int k = (int)(i % INPAD);
    Xbf[i] = (k < IN_DIM)
               ? f2bf(x[((long long)b * T_SEQ + t) * IN_DIM + k])
               : (unsigned short)0;
  }
  // h ping-pong buffers start at zero (h0 = 0)
  for (long long i = tid0; i < 2LL * B_SZ * HID; i += stride) hbuf[i] = 0;
  // MDN weight padding (zeros keep padded lanes exactly zero through relu)
  for (long long i = tid0; i < (long long)HPAD * HID; i += stride) {
    int r = (int)(i / HID), k = (int)(i % HID);
    W1p[i] = (r < 100) ? f2bf(W1[r * HID + k]) : (unsigned short)0;
  }
  for (long long i = tid0; i < (long long)HPAD * HPAD; i += stride) {
    int r = (int)(i / HPAD), k = (int)(i % HPAD);
    W2p[i] = (r < 100 && k < 100) ? f2bf(W2[r * 100 + k]) : (unsigned short)0;
  }
  for (long long i = tid0; i < (long long)OPAD * HPAD; i += stride) {
    int r = (int)(i / HPAD), k = (int)(i % HPAD);
    W3p[i] = (r < 170 && k < 100) ? f2bf(W3[r * 100 + k]) : (unsigned short)0;
  }
  for (long long i = tid0; i < HPAD; i += stride) {
    b1p[i] = (i < 100) ? b1[i] : 0.f;
    b2p[i] = (i < 100) ? b2[i] : 0.f;
  }
  for (long long i = tid0; i < OPAD; i += stride)
    b3p[i] = (i < 170) ? b3[i] : 0.f;
}

// ---------------------------------------------------------------------------
// Phase 1: cooperative LSTM scan. 32 workgroups x 8 waves = 256 waves.
// Wave (wg, w): hidden tile n = wg>>1, batch tile m = (wg&1)*8 + w.
// Per step: gates[m-tile, n-tile + g*256] for g=i,f,g,o  ->  c,h update.
// ---------------------------------------------------------------------------
__global__ void __launch_bounds__(256, 1) lstm_scan_kernel(
    const float* __restrict__ Wih,  // [1024, 35]
    const float* __restrict__ Whh,  // [1024, 256]
    const float* __restrict__ bih, const float* __restrict__ bhh,
    const unsigned short* __restrict__ Xbf,   // [TM1][B][INPAD]
    unsigned short* __restrict__ hbuf,        // 2 x [B][HID]
    unsigned short* __restrict__ Hall,        // [TM1][B][HID]
    unsigned* __restrict__ cnt)               // grid barrier counter (zeroed)
{
  __shared__ __align__(16) unsigned short sWhh[64 * HID];    // 32 KB
  __shared__ __align__(16) unsigned short sWih[64 * INPAD];  //  8 KB
  __shared__ float sBias[64];

  const int tid  = threadIdx.x;
  const int lane = tid & 31;
  const int wave = tid >> 5;
  const int nBase = (blockIdx.x >> 1) * 16;
  const int mBase = ((blockIdx.x & 1) * 8 + wave) * 16;

  // Stage this WG's 64 gate rows (4 gates x 16) of W_hh / W_ih into LDS (bf16).
  for (int i = tid; i < 64 * HID; i += 256) {
    int lr = i >> 8, k = i & 255;
    int j = (lr >> 4) * HID + nBase + (lr & 15);  // global gate row
    sWhh[i] = f2bf(Whh[j * HID + k]);
  }
  for (int i = tid; i < 64 * INPAD; i += 256) {
    int lr = i >> 6, k = i & 63;
    int j = (lr >> 4) * HID + nBase + (lr & 15);
    sWih[i] = (k < IN_DIM) ? f2bf(Wih[j * IN_DIM + k]) : (unsigned short)0;
  }
  if (tid < 64) {
    int j = (tid >> 4) * HID + nBase + (tid & 15);
    sBias[tid] = bih[j] + bhh[j];
  }
  __syncthreads();

  float bias4[4];
#pragma unroll
  for (int g = 0; g < 4; ++g) bias4[g] = sBias[g * 16 + (lane & 15)];

  const int arow    = mBase + (lane & 15);
  const int koff    = (lane >> 4) << 3;
  const int outcol  = nBase + (lane & 15);
  const int outrow0 = mBase + ((lane >> 4) << 3);

  v8f c = {};  // persistent cell state: c[i] = C[outrow0+i][outcol]

  for (int t = 0; t < TM1; ++t) {
    const unsigned short* hin  = hbuf + (size_t)(t & 1) * (B_SZ * HID);
    unsigned short*       hout = hbuf + (size_t)((t + 1) & 1) * (B_SZ * HID);
    const unsigned short* xt   = Xbf + (size_t)t * (B_SZ * INPAD);
    if (t + 1 < TM1)
      __builtin_prefetch(xt + (size_t)(B_SZ * INPAD) + arow * INPAD, 0, 0);

    v8f acc[4] = {{}, {}, {}, {}};
    // recurrent part: K = 256
#pragma unroll
    for (int kc = 0; kc < 8; ++kc) {
      const int k0 = kc * 32 + koff;
      v16bf a = ldfrag(hin + arow * HID + k0);
#pragma unroll
      for (int g = 0; g < 4; ++g) {
        v16bf b = ldfrag(&sWhh[(g * 16 + (lane & 15)) * HID + k0]);
        acc[g] = wmma_bf16(a, b, acc[g]);
      }
    }
    // input part: K = 64 (zero-padded 35)
#pragma unroll
    for (int kc = 0; kc < 2; ++kc) {
      const int k0 = kc * 32 + koff;
      v16bf a = ldfrag(xt + arow * INPAD + k0);
#pragma unroll
      for (int g = 0; g < 4; ++g) {
        v16bf b = ldfrag(&sWih[(g * 16 + (lane & 15)) * INPAD + k0]);
        acc[g] = wmma_bf16(a, b, acc[g]);
      }
    }
    // elementwise gate math + state update, write h (bf16) for next step
    unsigned short* hallrow = Hall + (size_t)t * (B_SZ * HID);
#pragma unroll
    for (int i = 0; i < 8; ++i) {
      float iv = sigm(acc[0][i] + bias4[0]);
      float fv = sigm(acc[1][i] + bias4[1]);
      float gv = tanhf(acc[2][i] + bias4[2]);
      float ov = sigm(acc[3][i] + bias4[3]);
      float cn = fv * c[i] + iv * gv;
      c[i] = cn;
      unsigned short hb = f2bf(ov * tanhf(cn));
      int row = outrow0 + i;
      hout[row * HID + outcol]    = hb;
      hallrow[row * HID + outcol] = hb;
    }
    // device-wide barrier (monotonic counter; double-buffered h needs one/step)
    __threadfence();
    __syncthreads();
    if (tid == 0) {
      __hip_atomic_fetch_add(cnt, 1u, __ATOMIC_ACQ_REL, __HIP_MEMORY_SCOPE_AGENT);
      const unsigned target = (unsigned)NBLK_SCAN * (unsigned)(t + 1);
      while (__hip_atomic_load(cnt, __ATOMIC_ACQUIRE,
                               __HIP_MEMORY_SCOPE_AGENT) < target) {
        __builtin_amdgcn_s_sleep(2);
      }
    }
    __syncthreads();
  }
}

// ---------------------------------------------------------------------------
// Phase 2: flat bf16 WMMA GEMMs, one 16x16 D-tile per wave.
//   out[M, Npad] = act(A[M, K] @ W[Npad, K]^T + bias)
// ---------------------------------------------------------------------------
__global__ void __launch_bounds__(256) gemm_bf16_kernel(
    const unsigned short* __restrict__ A, const unsigned short* __restrict__ W,
    const float* __restrict__ bias, unsigned short* __restrict__ out,
    int Mtiles, int Ntiles, int K)
{
  const int tile = blockIdx.x * 8 + (threadIdx.x >> 5);
  const int lane = threadIdx.x & 31;
  const int mt = tile / Ntiles, nt = tile % Ntiles;
  if (mt >= Mtiles) return;
  const int Npad = Ntiles * 16;
  const size_t arow = (size_t)(mt * 16 + (lane & 15)) * K;
  const size_t brow = (size_t)(nt * 16 + (lane & 15)) * K;
  const int koff = (lane >> 4) << 3;
  v8f acc = {};
  for (int k0 = koff; k0 < K; k0 += 32)
    acc = wmma_bf16(ldfrag(A + arow + k0), ldfrag(W + brow + k0), acc);
  const int n  = nt * 16 + (lane & 15);
  const int m0 = mt * 16 + ((lane >> 4) << 3);
  const float bv = bias[n];
#pragma unroll
  for (int i = 0; i < 8; ++i)
    out[(size_t)(m0 + i) * Npad + n] = f2bf(fmaxf(acc[i] + bv, 0.f));
}

__global__ void __launch_bounds__(256) gemm_f32_kernel(
    const unsigned short* __restrict__ A, const unsigned short* __restrict__ W,
    const float* __restrict__ bias, float* __restrict__ out,
    int Mtiles, int Ntiles, int K)
{
  const int tile = blockIdx.x * 8 + (threadIdx.x >> 5);
  const int lane = threadIdx.x & 31;
  const int mt = tile / Ntiles, nt = tile % Ntiles;
  if (mt >= Mtiles) return;
  const int Npad = Ntiles * 16;
  const size_t arow = (size_t)(mt * 16 + (lane & 15)) * K;
  const size_t brow = (size_t)(nt * 16 + (lane & 15)) * K;
  const int koff = (lane >> 4) << 3;
  v8f acc = {};
  for (int k0 = koff; k0 < K; k0 += 32)
    acc = wmma_bf16(ldfrag(A + arow + k0), ldfrag(W + brow + k0), acc);
  const int n  = nt * 16 + (lane & 15);
  const int m0 = mt * 16 + ((lane >> 4) << 3);
  const float bv = bias[n];
#pragma unroll
  for (int i = 0; i < 8; ++i)
    out[(size_t)(m0 + i) * Npad + n] = acc[i] + bv;
}

// ---------------------------------------------------------------------------
// Phase 3: softmax / exp / scatter to batch-first outputs.
// obuf row r = t*B + b; out = coeff[B,TM1,5] | mean[B,TM1,160] | var[B,TM1,5]
// ---------------------------------------------------------------------------
__global__ void __launch_bounds__(256) finalize_kernel(
    const float* __restrict__ obuf, float* __restrict__ out)
{
  const size_t OM = (size_t)B_SZ * TM1 * 5;
  const size_t OV = OM + (size_t)B_SZ * TM1 * 160;
  const long long stride = (long long)gridDim.x * blockDim.x;
  const long long tid0   = (long long)blockIdx.x * blockDim.x + threadIdx.x;

  for (long long r = tid0; r < MROWS; r += stride) {
    const float* o = obuf + (size_t)r * OPAD;
    const int t = (int)(r >> 8), b = (int)(r & 255);
    float mx = o[0];
#pragma unroll
    for (int k = 1; k < 5; ++k) mx = fmaxf(mx, o[k]);
    float e[5], s = 0.f;
#pragma unroll
    for (int k = 0; k < 5; ++k) { e[k] = __expf(o[k] - mx); s += e[k]; }
    const float inv = 1.0f / s;
    const size_t co = ((size_t)b * TM1 + t) * 5;
#pragma unroll
    for (int k = 0; k < 5; ++k) {
      out[co + k]      = e[k] * inv;         // coeff (softmax, TEMP=1)
      out[OV + co + k] = __expf(o[5 + k]);   // var = exp(.)*TEMP
    }
  }
  for (long long i = tid0; i < (long long)MROWS * 160; i += stride) {
    const long long r = i / 160;
    const int j = (int)(i % 160);
    const int t = (int)(r >> 8), b = (int)(r & 255);
    out[OM + ((size_t)b * TM1 + t) * 160 + j] = obuf[(size_t)r * OPAD + 10 + j];
  }
}

// ---------------------------------------------------------------------------
extern "C" void kernel_launch(void* const* d_in, const int* in_sizes, int n_in,
                              void* d_out, int out_size, void* d_ws, size_t ws_size,
                              hipStream_t stream) {
  const float* x    = (const float*)d_in[0];
  const float* Wih  = (const float*)d_in[1];
  const float* Whh  = (const float*)d_in[2];
  const float* bih  = (const float*)d_in[3];
  const float* bhh  = (const float*)d_in[4];
  const float* W1   = (const float*)d_in[5];
  const float* b1   = (const float*)d_in[6];
  const float* W2   = (const float*)d_in[7];
  const float* b2   = (const float*)d_in[8];
  const float* W3   = (const float*)d_in[9];
  const float* b3   = (const float*)d_in[10];
  float* out = (float*)d_out;

  char* ws = (char*)d_ws;
  size_t off = 0;
  auto alloc = [&](size_t bytes) -> void* {
    void* p = ws + off;
    off = (off + bytes + 255) & ~(size_t)255;
    return p;
  };
  unsigned*       cnt  = (unsigned*)alloc(256);
  unsigned short* Xbf  = (unsigned short*)alloc((size_t)TM1 * B_SZ * INPAD * 2);
  unsigned short* hbuf = (unsigned short*)alloc(2LL * B_SZ * HID * 2);
  unsigned short* Hall = (unsigned short*)alloc((size_t)TM1 * B_SZ * HID * 2);
  unsigned short* W1p  = (unsigned short*)alloc((size_t)HPAD * HID * 2);
  unsigned short* W2p  = (unsigned short*)alloc((size_t)HPAD * HPAD * 2);
  unsigned short* W3p  = (unsigned short*)alloc((size_t)OPAD * HPAD * 2);
  float*          b1p  = (float*)alloc(HPAD * 4);
  float*          b2p  = (float*)alloc(HPAD * 4);
  float*          b3p  = (float*)alloc(OPAD * 4);
  unsigned short* hid1 = (unsigned short*)alloc((size_t)MROWS * HPAD * 2);
  unsigned short* hid2 = (unsigned short*)alloc((size_t)MROWS * HPAD * 2);
  float*          obuf = (float*)alloc((size_t)MROWS * OPAD * 4);

  hipMemsetAsync(cnt, 0, 256, stream);

  prep_kernel<<<2048, 256, 0, stream>>>(x, W1, b1, W2, b2, W3, b3,
                                        Xbf, hbuf, W1p, W2p, W3p, b1p, b2p, b3p);

  lstm_scan_kernel<<<NBLK_SCAN, 256, 0, stream>>>(Wih, Whh, bih, bhh,
                                                  Xbf, hbuf, Hall, cnt);

  const int Mtiles = MROWS / 16;                       // 16368
  gemm_bf16_kernel<<<(Mtiles * 8 + 7) / 8, 256, 0, stream>>>(
      Hall, W1p, b1p, hid1, Mtiles, HPAD / 16, HID);   // K=256, N=128
  gemm_bf16_kernel<<<(Mtiles * 8 + 7) / 8, 256, 0, stream>>>(
      hid1, W2p, b2p, hid2, Mtiles, HPAD / 16, HPAD);  // K=128, N=128
  gemm_f32_kernel<<<(Mtiles * 11 + 7) / 8, 256, 0, stream>>>(
      hid2, W3p, b3p, obuf, Mtiles, OPAD / 16, HPAD);  // K=128, N=176

  finalize_kernel<<<4096, 256, 0, stream>>>(obuf, out);
}
